// Attention_35278861369683
// MI455X (gfx1250) — matrix operand order repaired and checked by reading, not compiled, not run on hardware
//
#include <hip/hip_runtime.h>
#include <hip/hip_bf16.h>

typedef __bf16 v16bf __attribute__((ext_vector_type(16)));
typedef float  v8f   __attribute__((ext_vector_type(8)));
typedef int    v4i_async __attribute__((vector_size(16)));

#if defined(__AMDGCN__) && __has_builtin(__builtin_amdgcn_global_load_async_to_lds_b128) && __has_builtin(__builtin_amdgcn_s_wait_asynccnt)
#define ASYNC_OK 1
typedef __attribute__((address_space(1))) v4i_async* gbl_v4i_p;
typedef __attribute__((address_space(3))) v4i_async* lds_v4i_p;
#else
#define ASYNC_OK 0
#endif

// Copy 16 bytes global->LDS, async (CDNA5 GLOBAL_LOAD_ASYNC_TO_LDS_B128) when
// the builtin exists, else a plain vector copy.
static __device__ __forceinline__ void cp16_g2l(__bf16* dst, const __bf16* src) {
#if ASYNC_OK
    gbl_v4i_p gp = (gbl_v4i_p)(uintptr_t)src;
    lds_v4i_p lp = (lds_v4i_p)(unsigned int)(uintptr_t)dst;  // generic LDS VA -> 32-bit LDS offset
    __builtin_amdgcn_global_load_async_to_lds_b128(gp, lp, 0, 0);
#else
    *(uint4*)dst = *(const uint4*)src;
#endif
}

static __device__ __forceinline__ void async_join() {
#if ASYNC_OK
    __builtin_amdgcn_s_wait_asynccnt(0);
#endif
}

static __device__ __forceinline__ v8f wmma_bf16(v16bf a, v16bf b, v8f c) {
    return __builtin_amdgcn_wmma_f32_16x16x32_bf16(false, a, false, b, (short)0, c,
                                                   false, false);
}

// A-fragment (16x32 MxK) from row-major [M][K] tile: lane = row (lane&15);
// lanes<16 hold K {0..7,16..23}, lanes>=16 hold K {8..15,24..31}.
// Two contiguous 16B runs per lane -> merges to 2x ds_load_b128.
static __device__ __forceinline__ v16bf load_a_frag(const __bf16* tile, int stride, int lane) {
    int row = lane & 15;
    int kb  = (lane >> 4) << 3;
    const __bf16* p = tile + row * stride + kb;
    v16bf a;
#pragma unroll
    for (int e = 0; e < 8; ++e) a[e] = p[e];
#pragma unroll
    for (int e = 0; e < 8; ++e) a[8 + e] = p[16 + e];
    return a;
}

// B-fragment (32x16 KxN) where the tile is stored as [N][K] (transposed):
// lane = col n (lane&15);
// lanes<16 hold K {0..7,16..23}, lanes>=16 hold K {8..15,24..31}. Merges to 2x ds_load_b128.
static __device__ __forceinline__ v16bf load_row_frag(const __bf16* tile, int stride,
                                                      int n0, int k0, int lane) {
    int n  = n0 + (lane & 15);
    int kb = k0 + ((lane >> 4) << 3);
    const __bf16* p = tile + n * stride + kb;
    v16bf b;
#pragma unroll
    for (int e = 0; e < 8; ++e) b[e] = p[e];
#pragma unroll
    for (int e = 0; e < 8; ++e) b[8 + e] = p[16 + e];
    return b;
}

__global__ void f32_to_bf16(const float* __restrict__ in, __bf16* __restrict__ out, int n) {
    for (int i = blockIdx.x * blockDim.x + threadIdx.x; i < n; i += gridDim.x * blockDim.x)
        out[i] = (__bf16)in[i];
}

// --------------------------- shared GEMM pieces ----------------------------
// Block tile 128x128, BK=32, 8 waves (4M x 2N), each wave 32x64 = 2x4 frags.
static __device__ __forceinline__ void gemm_stage(
        const __bf16* __restrict__ X, const __bf16* __restrict__ W,
        int Kd, int Nd, __bf16* As, __bf16* Bst,
        int m0g, int n0g, int k0, int tid) {
    // A tile: row-major, async DMA into LDS
    int arow = tid >> 1, acol = (tid & 1) * 16;
    const __bf16* ga = X + (size_t)(m0g + arow) * Kd + k0 + acol;
    cp16_g2l(As + arow * 40 + acol,     ga);
    cp16_g2l(As + arow * 40 + acol + 8, ga + 8);
    // B tile: coalesced row loads of W, scatter transposed [N][K] into LDS
    int brow = tid & 31, bc0 = (tid >> 5) * 16;
    uint4 wrow[2];
    wrow[0] = *(const uint4*)(W + (size_t)(k0 + brow) * Nd + n0g + bc0);
    wrow[1] = *(const uint4*)(W + (size_t)(k0 + brow) * Nd + n0g + bc0 + 8);
    const __bf16* wp = (const __bf16*)wrow;
#pragma unroll
    for (int e = 0; e < 16; ++e) Bst[(bc0 + e) * 40 + brow] = wp[e];
}

static __device__ __forceinline__ void gemm_step(
        const __bf16* As, const __bf16* Bst,
        int wm0, int wn0, int lane, v8f acc[2][4]) {
    v16bf af0 = load_a_frag(As + (wm0)      * 40, 40, lane);
    v16bf af1 = load_a_frag(As + (wm0 + 16) * 40, 40, lane);
#pragma unroll
    for (int j = 0; j < 4; ++j) {
        v16bf bf = load_row_frag(Bst, 40, wn0 + j * 16, 0, lane);
        acc[0][j] = wmma_bf16(af0, bf, acc[0][j]);
        acc[1][j] = wmma_bf16(af1, bf, acc[1][j]);
    }
}

// ---------------------------------------------------------------------------
// Kernel 1: qkv = Xbf @ Wqkvbf (8192 x 1024 x 3072) with fused RoPE on q/k and
// softmax scale folded into q. Double-buffered async pipeline.
// ---------------------------------------------------------------------------
__global__ __launch_bounds__(256) void qkv_rope_kernel(
        const __bf16* __restrict__ X, const __bf16* __restrict__ W,
        __bf16* __restrict__ Qo, __bf16* __restrict__ Ko, __bf16* __restrict__ Vo) {
    constexpr int Kd = 1024, Nd = 3072;
    __shared__ __bf16 As[2][128 * 40];    // [M][K] row-major
    __shared__ __bf16 Bst[2][128 * 40];   // [N][K] transposed

    int tid  = threadIdx.x;
    int lane = tid & 31, wv = tid >> 5;
    int m0g  = blockIdx.x * 128;
    int n0g  = blockIdx.y * 128;
    int wm0  = (wv & 3) * 32;
    int wn0  = (wv >> 2) * 64;

    v8f acc[2][4];
#pragma unroll
    for (int i = 0; i < 2; ++i)
#pragma unroll
        for (int j = 0; j < 4; ++j)
#pragma unroll
            for (int r = 0; r < 8; ++r) acc[i][j][r] = 0.0f;

    gemm_stage(X, W, Kd, Nd, As[0], Bst[0], m0g, n0g, 0, tid);
    async_join();
    __syncthreads();

    constexpr int NK = Kd / 32;
    for (int i = 0; i < NK; ++i) {
        int cur = i & 1, nxt = cur ^ 1;
        if (i + 1 < NK)   // prefetch next tile while current computes
            gemm_stage(X, W, Kd, Nd, As[nxt], Bst[nxt], m0g, n0g, (i + 1) * 32, tid);
        gemm_step(As[cur], Bst[cur], wm0, wn0, lane, acc);
        async_join();
        __syncthreads();
    }

    int halfl = lane >> 4, ln = lane & 15;
#pragma unroll
    for (int i = 0; i < 2; ++i) {
#pragma unroll
        for (int j = 0; j < 4; ++j) {
            int colg = n0g + wn0 + j * 16 + ln;  // 16-col window: one section, one head
            int sec  = colg >> 10;
            int rem  = colg & 1023;
            int head = rem >> 6;
            int d    = rem & 63;
            int ii   = d >> 1;
            // theta = 10000^(-2*(i-1)/64) = exp(-ln(1e4)*2*(i-1)/64)
            float theta = __expf(-0.2878231366f * ((float)ii - 1.0f));
#pragma unroll
            for (int r = 0; r < 8; ++r) {
                int rowg = m0g + wm0 + i * 16 + r + halfl * 8;
                int bidx = rowg >> 11;
                int srow = rowg & 2047;
                float v = acc[i][j][r];
                float outv;
                if (sec == 2) {
                    outv = v;                         // v passthrough
                } else {                              // RoPE: pair cols = adjacent lanes
                    float partner = __shfl_xor(v, 1);
                    float ang = theta * (float)srow;
                    float sn, cs;
                    __sincosf(ang, &sn, &cs);
                    outv = ((d & 1) == 0) ? (cs * v - sn * partner)
                                          : (sn * partner + cs * v);
                    if (sec == 0) outv *= 0.125f;     // fold 1/sqrt(64) into q
                }
                __bf16* dst = (sec == 0) ? Qo : (sec == 1) ? Ko : Vo;
                dst[((size_t)(bidx * 16 + head) * 2048 + srow) * 64 + d] = (__bf16)outv;
            }
        }
    }
}

// ---------------------------------------------------------------------------
// Kernel 2: causal flash attention. grid = (S/64, B*H), 4 waves per block.
// Double-buffered K/V staging; Q in registers; online softmax; P restaged via
// per-wave LDS (per-wave DS ordering).
// ---------------------------------------------------------------------------
__global__ __launch_bounds__(128) void flash_kernel(
        const __bf16* __restrict__ Q, const __bf16* __restrict__ Kt,
        const __bf16* __restrict__ V, __bf16* __restrict__ ctx) {
    constexpr int LDK = 72, LDVT = 72, LDP = 40;
    __shared__ __bf16 Ks[2][64 * LDK];    // [key][d]
    __shared__ __bf16 Vt[2][64 * LDVT];   // [d][key]
    __shared__ __bf16 Ps[4][16 * LDP];

    int qb = blockIdx.x;
    int bh = blockIdx.y;
    int b  = bh >> 4, h = bh & 15;
    int tid = threadIdx.x, wv = tid >> 5, lane = tid & 31;
    int ln = lane & 15, halfl = lane >> 4;

    const __bf16* Qp = Q  + (size_t)bh * 2048 * 64;
    const __bf16* Kp = Kt + (size_t)bh * 2048 * 64;
    const __bf16* Vp = V  + (size_t)bh * 2048 * 64;
    int qr0 = qb * 64 + wv * 16;

    v16bf aq[2];
#pragma unroll
    for (int kk = 0; kk < 2; ++kk) {
        const __bf16* qrow = Qp + (size_t)(qr0 + ln) * 64 + kk * 32 + halfl * 8;
#pragma unroll
        for (int e = 0; e < 8; ++e) aq[kk][e] = qrow[e];
#pragma unroll
        for (int e = 0; e < 8; ++e) aq[kk][8 + e] = qrow[16 + e];
    }

    v8f acc[4];
    float m_r[8], l_r[8];
#pragma unroll
    for (int r = 0; r < 8; ++r) { m_r[r] = -3.0e38f; l_r[r] = 0.0f; }
#pragma unroll
    for (int dn = 0; dn < 4; ++dn)
#pragma unroll
        for (int r = 0; r < 8; ++r) acc[dn][r] = 0.0f;

    int krow = tid >> 1;          // 0..63
    int kc0  = (tid & 1) * 32;    // 0 or 32

    // stage K/V tile for block j into buffer buf
    auto stage_kv = [&](int j, int buf) {
        const __bf16* gk = Kp + (size_t)(j * 64 + krow) * 64 + kc0;
        cp16_g2l(Ks[buf] + krow * LDK + kc0,     gk);      // K row-major, async
        cp16_g2l(Ks[buf] + krow * LDK + kc0 + 8, gk + 8);
        uint4 vv[4];                                        // V: read row, scatter transposed
        const uint4* gv = (const uint4*)(Vp + (size_t)(j * 64 + krow) * 64 + kc0);
#pragma unroll
        for (int t = 0; t < 4; ++t) vv[t] = gv[t];
        const __bf16* vp = (const __bf16*)vv;
#pragma unroll
        for (int e = 0; e < 32; ++e) Vt[buf][(kc0 + e) * LDVT + krow] = vp[e];
    };

    stage_kv(0, 0);
    async_join();
    __syncthreads();

    for (int j = 0; j <= qb; ++j) {
        int cur = j & 1, nxt = cur ^ 1;
        if (j < qb) stage_kv(j + 1, nxt);   // prefetch next K/V while computing

#pragma unroll
        for (int kk = 0; kk < 2; ++kk) {
            int kbase = kk * 32;
            v8f s0, s1;
#pragma unroll
            for (int r = 0; r < 8; ++r) { s0[r] = 0.0f; s1[r] = 0.0f; }

            v16bf bt;
            bt = load_row_frag(Ks[cur], LDK, kbase,      0,  lane); s0 = wmma_bf16(aq[0], bt, s0);
            bt = load_row_frag(Ks[cur], LDK, kbase,      32, lane); s0 = wmma_bf16(aq[1], bt, s0);
            bt = load_row_frag(Ks[cur], LDK, kbase + 16, 0,  lane); s1 = wmma_bf16(aq[0], bt, s1);
            bt = load_row_frag(Ks[cur], LDK, kbase + 16, 32, lane); s1 = wmma_bf16(aq[1], bt, s1);

            if (j == qb) {   // causal mask on diagonal block
#pragma unroll
                for (int r = 0; r < 8; ++r) {
                    int rowg = qr0 + r + halfl * 8;
                    int c0   = j * 64 + kbase + ln;
                    if (c0 > rowg)      s0[r] = -3.0e38f;
                    if (c0 + 16 > rowg) s1[r] = -3.0e38f;
                }
            }

            __bf16* pw = Ps[wv];
#pragma unroll
            for (int r = 0; r < 8; ++r) {
                float mx = fmaxf(s0[r], s1[r]);
#pragma unroll
                for (int off = 1; off < 16; off <<= 1) mx = fmaxf(mx, __shfl_xor(mx, off));
                float mn = fmaxf(m_r[r], mx);
                float alpha = __expf(m_r[r] - mn);
                float p0 = __expf(s0[r] - mn);
                float p1 = __expf(s1[r] - mn);
                float rs = p0 + p1;
#pragma unroll
                for (int off = 1; off < 16; off <<= 1) rs += __shfl_xor(rs, off);
                l_r[r] = l_r[r] * alpha + rs;
                m_r[r] = mn;
#pragma unroll
                for (int dn = 0; dn < 4; ++dn) acc[dn][r] *= alpha;
                pw[(r + halfl * 8) * LDP + ln]      = (__bf16)p0;
                pw[(r + halfl * 8) * LDP + 16 + ln] = (__bf16)p1;
            }

            v16bf pa = load_a_frag(pw, LDP, lane);
#pragma unroll
            for (int dn = 0; dn < 4; ++dn) {
                v16bf bv = load_row_frag(Vt[cur], LDVT, dn * 16, kbase, lane);
                acc[dn] = wmma_bf16(pa, bv, acc[dn]);
            }
        }
        async_join();
        __syncthreads();
    }

#pragma unroll
    for (int r = 0; r < 8; ++r) {
        int srow = qr0 + r + halfl * 8;
        float invl = 1.0f / l_r[r];
#pragma unroll
        for (int dn = 0; dn < 4; ++dn) {
            int d = dn * 16 + ln;
            ctx[((size_t)b * 2048 + srow) * 1024 + h * 64 + d] =
                (__bf16)(acc[dn][r] * invl);
        }
    }
}

// ---------------------------------------------------------------------------
// Kernel 3: out = ctx @ Wout + bout (8192 x 1024 x 1024), fp32 output.
// Same pipelined tiling as kernel 1.
// ---------------------------------------------------------------------------
__global__ __launch_bounds__(256) void out_proj_kernel(
        const __bf16* __restrict__ X, const __bf16* __restrict__ W,
        const float* __restrict__ bias, float* __restrict__ out) {
    constexpr int Kd = 1024, Nd = 1024;
    __shared__ __bf16 As[2][128 * 40];
    __shared__ __bf16 Bst[2][128 * 40];

    int tid  = threadIdx.x;
    int lane = tid & 31, wv = tid >> 5;
    int m0g  = blockIdx.x * 128;
    int n0g  = blockIdx.y * 128;
    int wm0  = (wv & 3) * 32;
    int wn0  = (wv >> 2) * 64;

    v8f acc[2][4];
#pragma unroll
    for (int i = 0; i < 2; ++i)
#pragma unroll
        for (int j = 0; j < 4; ++j)
#pragma unroll
            for (int r = 0; r < 8; ++r) acc[i][j][r] = 0.0f;

    gemm_stage(X, W, Kd, Nd, As[0], Bst[0], m0g, n0g, 0, tid);
    async_join();
    __syncthreads();

    constexpr int NK = Kd / 32;
    for (int i = 0; i < NK; ++i) {
        int cur = i & 1, nxt = cur ^ 1;
        if (i + 1 < NK)
            gemm_stage(X, W, Kd, Nd, As[nxt], Bst[nxt], m0g, n0g, (i + 1) * 32, tid);
        gemm_step(As[cur], Bst[cur], wm0, wn0, lane, acc);
        async_join();
        __syncthreads();
    }

    int halfl = lane >> 4, ln = lane & 15;
#pragma unroll
    for (int i = 0; i < 2; ++i) {
#pragma unroll
        for (int j = 0; j < 4; ++j) {
            int colg = n0g + wn0 + j * 16 + ln;
            float bv = bias[colg];
#pragma unroll
            for (int r = 0; r < 8; ++r) {
                int rowg = m0g + wm0 + i * 16 + r + halfl * 8;
                out[(size_t)rowg * Nd + colg] = acc[i][j][r] + bv;
            }
        }
    }
}

extern "C" void kernel_launch(void* const* d_in, const int* in_sizes, int n_in,
                              void* d_out, int out_size, void* d_ws, size_t ws_size,
                              hipStream_t stream) {
    const float* x    = (const float*)d_in[0];
    const float* Wqkv = (const float*)d_in[1];
    const float* Wout = (const float*)d_in[2];
    const float* bout = (const float*)d_in[3];

    char* ws = (char*)d_ws;
    __bf16* xb    = (__bf16*)(ws + 0);          // 16 MB
    __bf16* wqkvb = (__bf16*)(ws + 16777216);   // 6 MB
    __bf16* woutb = (__bf16*)(ws + 23068672);   // 2 MB
    __bf16* qbuf  = (__bf16*)(ws + 25165824);   // 16 MB  [B,H,S,64]
    __bf16* kbuf  = (__bf16*)(ws + 41943040);   // 16 MB
    __bf16* vbuf  = (__bf16*)(ws + 58720256);   // 16 MB
    __bf16* ctx   = (__bf16*)(ws + 75497472);   // 16 MB  [B,S,1024]

    f32_to_bf16<<<2048, 256, 0, stream>>>(x,    xb,    4 * 2048 * 1024);
    f32_to_bf16<<<2048, 256, 0, stream>>>(Wqkv, wqkvb, 1024 * 3072);
    f32_to_bf16<<<1024, 256, 0, stream>>>(Wout, woutb, 1024 * 1024);

    dim3 g1(64, 24);
    qkv_rope_kernel<<<g1, 256, 0, stream>>>(xb, wqkvb, qbuf, kbuf, vbuf);

    dim3 g2(32, 64);
    flash_kernel<<<g2, 128, 0, stream>>>(qbuf, kbuf, vbuf, ctx);

    dim3 g3(64, 8);
    out_proj_kernel<<<g3, 256, 0, stream>>>(ctx, woutb, bout, (float*)d_out);
}
